// HiPPO_LegT_77266461655040
// MI455X (gfx1250) — matrix operand order repaired and checked by reading, not compiled, not run on hardware
//
#include <hip/hip_runtime.h>
#include <hip/hip_bf16.h>
#include <stdint.h>

#define LEN   4096
#define NST   256
#define HREV  4168                 // 4096 reversed h + 72 zero tail
#define CPYS  4175                 // element stride between LDS copies (== -1 mod 8)

typedef __attribute__((ext_vector_type(16))) __bf16 v16bf;
typedef __attribute__((ext_vector_type(8)))  float  v8f;
typedef __attribute__((ext_vector_type(4)))  unsigned int tdm_v4u;
typedef __attribute__((ext_vector_type(8)))  int          tdm_v8i;
typedef __attribute__((ext_vector_type(4)))  int          tdm_v4i;

__device__ __forceinline__ uint16_t f32_to_bf16_bits(float f) {
    union { float f; uint32_t u; } c; c.f = f;
    uint32_t u = c.u;
    uint32_t r = u + 0x7FFFu + ((u >> 16) & 1u);   // round-to-nearest-even
    return (uint16_t)(r >> 16);
}

// ---- Step 1a: W[k] = C^T A^k for k = 0..63 (single WG, 64 serial matvecs) ----
__global__ void seq_w_init(const float* __restrict__ A,
                           const float* __restrict__ C,
                           float* __restrict__ W) {
    __shared__ float w[NST];
    int j = threadIdx.x;
    w[j] = C[j];
    __syncthreads();
    for (int k = 0; k < 64; ++k) {
        W[k * NST + j] = w[j];
        float acc = 0.f;
        for (int i = 0; i < NST; ++i)
            acc = fmaf(w[i], A[i * NST + j], acc);   // w_next = w @ A
        __syncthreads();
        w[j] = acc;
        __syncthreads();
    }
}

// ---- Step 1b: fp32 GEMM, C[M x 256] = A[M x 256] @ B[256 x 256] (squaring chain) ----
__global__ void gemm_f32_n256(const float* __restrict__ A,
                              const float* __restrict__ B,
                              float* __restrict__ C) {
    int col = blockIdx.x * 16 + threadIdx.x;
    int row = blockIdx.y * 16 + threadIdx.y;
    float acc = 0.f;
    const float* a = A + (size_t)row * NST;
    for (int k = 0; k < NST; ++k)
        acc = fmaf(a[k], B[k * NST + col], acc);
    C[(size_t)row * NST + col] = acc;
}

// ---- Step 1c: fused doubling level: Wdst[0:K) = W[0:K) @ M  and  Sq = M @ M ----
__global__ void level_gemm(const float* __restrict__ W,
                           const float* __restrict__ M,
                           float* __restrict__ Wdst,
                           float* __restrict__ Sq,
                           int K16) {
    int col = blockIdx.x * 16 + threadIdx.x;
    int by  = blockIdx.y;
    const float* arow;
    float* out;
    if (by < K16) {
        int row = by * 16 + threadIdx.y;
        arow = W    + (size_t)row * NST;
        out  = Wdst + (size_t)row * NST;
    } else {
        int row = (by - K16) * 16 + threadIdx.y;
        arow = M  + (size_t)row * NST;
        out  = Sq + (size_t)row * NST;
    }
    float acc = 0.f;
    for (int k = 0; k < NST; ++k)
        acc = fmaf(arow[k], M[k * NST + col], acc);
    out[col] = acc;
}

// ---- Step 2 (fused): hrev[4095-k] = W[k].B (bf16, reversed, zero tail) ; Xb = bf16(X) ----
__global__ void prep_h_x(const float* __restrict__ W,
                         const float* __restrict__ Bv,
                         uint16_t* __restrict__ hrev,
                         const float* __restrict__ X,
                         uint16_t* __restrict__ Xb) {
    int b = blockIdx.x;
    if (b < 512) {
        int wave = threadIdx.x >> 5;
        int lane = threadIdx.x & 31;
        int row  = b * 8 + wave;
        const float* wr = W + (size_t)row * NST;
        float acc = 0.f;
        for (int i = lane; i < NST; i += 32)
            acc = fmaf(wr[i], Bv[i], acc);
        for (int off = 16; off > 0; off >>= 1)
            acc += __shfl_down(acc, off, 32);
        if (lane == 0) hrev[(LEN - 1) - row] = f32_to_bf16_bits(acc);
        if (b == 0 && threadIdx.x < (HREV - LEN)) hrev[LEN + threadIdx.x] = 0;
    } else {
        int i = (b - 512) * 256 + threadIdx.x;
        Xb[i] = f32_to_bf16_bits(X[i]);
    }
}

// ---- Step 3: causal Toeplitz GEMM on WMMA bf16 ----
// One WG per 16-wide t-tile; 8 waves x (2 M-tiles) cover all 256 batch rows.
// LDS holds 8 alignment-staggered copies of hrev -> every lane's B fragment is
// one 16B-aligned 32-byte run => 2x ds_load_b128; A fragment => 2x global b128.
__global__ void __launch_bounds__(256)
conv_wmma(const uint16_t* __restrict__ Xb,
          const uint16_t* __restrict__ hrev,
          const float* __restrict__ X32,
          const float* __restrict__ Dp,
          float* __restrict__ Y) {
    extern __shared__ uint16_t hrs[];          // 8 * CPYS elements, base 16B aligned
    const int t0 = blockIdx.x * 16;

#if __has_builtin(__builtin_amdgcn_tensor_load_to_lds)
    if (threadIdx.x < 32) {                    // wave 0 drives the TDM
        const uint32_t lds0 = (uint32_t)(uintptr_t)hrs;
        const uint64_t ga   = (uint64_t)(uintptr_t)hrev;
        for (int r = 0; r < 8; ++r) {
            tdm_v4u g0;
            g0[0] = 1u;                                            // count=1
            g0[1] = lds0 + (uint32_t)r * (CPYS * 2u);              // lds_addr (bytes)
            g0[2] = (uint32_t)ga;                                  // global_addr lo
            g0[3] = (uint32_t)((ga >> 32) & 0x1FFFFFFull) | 0x80000000u; // addr hi | type=2
            tdm_v8i g1;
            g1[0] = 0;                                             // 1B elems, no flags
            g1[1] = (int)((HREV * 2u & 0xFFFFu) << 16);            // tensor_dim0 lo16
            g1[2] = (int)(((HREV * 2u) >> 16) | (1u << 16));       // tensor_dim0 hi | tensor_dim1=1
            g1[3] = (int)((HREV * 2u) << 16);                      // tile_dim0 = 8336 B
            g1[4] = 1;                                             // tile_dim1 = 1
            g1[5] = (int)(HREV * 2u);                              // tensor_dim0_stride
            g1[6] = 0; g1[7] = 0;
            tdm_v4i z4 = {0, 0, 0, 0};
            tdm_v8i z8 = {0, 0, 0, 0, 0, 0, 0, 0};
            __builtin_amdgcn_tensor_load_to_lds(g0, g1, z4, z4, z8, 0);
        }
        __builtin_amdgcn_s_wait_tensorcnt(0);
    }
    __syncthreads();
#else
    for (int e = threadIdx.x; e < 8 * CPYS; e += 256) {
        int r = e / CPYS, off = e - r * CPYS;
        hrs[e] = (off < HREV) ? hrev[off] : (uint16_t)0;
    }
    __syncthreads();
#endif

    const int lane = threadIdx.x & 31;
    const int wave = threadIdx.x >> 5;
    const int n    = lane & 15;                // t column within tile
    const int hi   = lane >> 4;
    const int kh   = hi * 16;                  // B-fragment K base (lanes 16-31 -> K 16..31)
    const int kb   = hi * 16;                  // A-fragment K base
    const int m0   = wave * 32;

    const int r0 = (23 - n) & 7;               // == jb mod 8, fixed per lane
    const uint16_t* hp = hrs + r0 * CPYS;      // staggered copy -> hp+jb is 16B aligned

    v8f acc0 = {}; v8f acc1 = {};

    const uint16_t* x0 = Xb + (size_t)(m0 +      n) * LEN;
    const uint16_t* x1 = Xb + (size_t)(m0 + 16 + n) * LEN;   // n == lane&15 == A-row M

    int jb = (LEN - 1) - t0 - n + kh;          // hr index of B-frag element 0 at s0=0

    for (int s0 = 0; s0 < t0 + 16; s0 += 32, jb += 32) {
        union { v16bf v; uint4 q[2]; } b;
        b.q[0] = *(const uint4*)(hp + jb);
        b.q[1] = *(const uint4*)(hp + jb + 8);

        union { v16bf v; uint4 q[2]; } a0, a1;
        const uint16_t* xp0 = x0 + s0 + kb;
        const uint16_t* xp1 = x1 + s0 + kb;
        a0.q[0] = *(const uint4*)(xp0);
        a0.q[1] = *(const uint4*)(xp0 + 8);
        a1.q[0] = *(const uint4*)(xp1);
        a1.q[1] = *(const uint4*)(xp1 + 8);

        acc0 = __builtin_amdgcn_wmma_f32_16x16x32_bf16(false, a0.v, false, b.v,
                                                       (short)0, acc0, false, false);
        acc1 = __builtin_amdgcn_wmma_f32_16x16x32_bf16(false, a1.v, false, b.v,
                                                       (short)0, acc1, false, false);
    }

    const float D = Dp[0];
    const int t = t0 + n;
    const float fac = (t == 0) ? 1.0f : 2.0f;
    #pragma unroll
    for (int r = 0; r < 8; ++r) {
        int br0 = m0 + r + 8 * hi;             // C/D layout: M = r + 8*(lane>>4)
        int br1 = br0 + 16;
        Y[(size_t)br0 * LEN + t] = fac * acc0[r] + D * X32[(size_t)br0 * LEN + t];
        Y[(size_t)br1 * LEN + t] = fac * acc1[r] + D * X32[(size_t)br1 * LEN + t];
    }
}

extern "C" void kernel_launch(void* const* d_in, const int* in_sizes, int n_in,
                              void* d_out, int out_size, void* d_ws, size_t ws_size,
                              hipStream_t stream) {
    (void)in_sizes; (void)n_in; (void)out_size; (void)ws_size;
    const float* X  = (const float*)d_in[0];   // (256, 4096)
    const float* A  = (const float*)d_in[1];   // (256, 256)
    const float* Bv = (const float*)d_in[2];   // (256,)
    const float* Cd = (const float*)d_in[3];   // (256,)
    const float* Dd = (const float*)d_in[4];   // (1,)
    float* Y = (float*)d_out;                  // (256, 4096)

    char* ws = (char*)d_ws;                    // ~6.8 MB used
    float*    W    = (float*)(ws);                         // 4096*256*4 = 4,194,304
    float*    P0   = (float*)(ws + 4194304);               // 262,144
    float*    P1   = (float*)(ws + 4194304 + 262144);      // 262,144
    uint16_t* hrev = (uint16_t*)(ws + 4194304 + 2*262144); // 8,352 (HREV*2 rounded)
    uint16_t* Xb   = (uint16_t*)(ws + 4194304 + 2*262144 + 8352); // 2,097,152

    dim3 b16(16, 16);

    hipLaunchKernelGGL(seq_w_init, dim3(1), dim3(256), 0, stream, A, Cd, W);

    // A^64 via repeated squaring
    hipLaunchKernelGGL(gemm_f32_n256, dim3(16, 16), b16, 0, stream, A,  A,  P0); // A^2
    hipLaunchKernelGGL(gemm_f32_n256, dim3(16, 16), b16, 0, stream, P0, P0, P1); // A^4
    hipLaunchKernelGGL(gemm_f32_n256, dim3(16, 16), b16, 0, stream, P1, P1, P0); // A^8
    hipLaunchKernelGGL(gemm_f32_n256, dim3(16, 16), b16, 0, stream, P0, P0, P1); // A^16
    hipLaunchKernelGGL(gemm_f32_n256, dim3(16, 16), b16, 0, stream, P1, P1, P0); // A^32
    hipLaunchKernelGGL(gemm_f32_n256, dim3(16, 16), b16, 0, stream, P0, P0, P1); // A^64

    // Fused doubling levels: W[K:2K)=W[0:K)@M ; M'=M@M (same launch)
    float* Mp = P1; float* Sp = P0;
    for (int K = 64; K <= 2048; K <<= 1) {
        int K16 = K / 16;
        int doSq = (K < 2048) ? 1 : 0;
        hipLaunchKernelGGL(level_gemm, dim3(16, K16 + (doSq ? 16 : 0)), b16, 0, stream,
                           W, Mp, W + (size_t)K * NST, Sp, K16);
        if (doSq) { float* t = Mp; Mp = Sp; Sp = t; }
    }

    hipLaunchKernelGGL(prep_h_x, dim3(512 + 4096), dim3(256), 0, stream, W, Bv, hrev, X, Xb);
    hipLaunchKernelGGL(conv_wmma, dim3(256), dim3(256), 8 * CPYS * 2 + 16, stream,
                       Xb, hrev, X, Dd, Y);
}
